// Enc_18373870092831
// MI455X (gfx1250) — compile-verified, matrix operand order checked by reading, not run on hardware
//
#include <hip/hip_runtime.h>

typedef __bf16 bf16_t;
typedef __attribute__((ext_vector_type(16))) __bf16 v16bf;
typedef __attribute__((ext_vector_type(8)))  float  v8f;

// round-to-nearest-even f32->bf16, two at a time, packed into one dword
__device__ inline unsigned pack_bf2(float lo, float hi) {
    union { float f; unsigned u; } a, b; a.f = lo; b.f = hi;
    unsigned ul = a.u + 0x7FFFu + ((a.u >> 16) & 1u);
    unsigned uh = b.u + 0x7FFFu + ((b.u >> 16) & 1u);
    return (ul >> 16) | (uh & 0xFFFF0000u);
}

__device__ inline int clampi(int v, int hi) { return v < 0 ? 0 : (v > hi ? hi : v); }

// K-index of fragment element e for this lane, per CDNA5 16-bit A/B layout:
// lanes 0-15: K base 0; lanes 16-31: K base 8; elems 0-7 -> base..base+7,
// elems 8-15 -> base+16..base+23.  (contiguous 16B runs -> ds_load_b128)
__device__ inline int kfrag(int lane, int e) {
    return ((lane & 16) ? 8 : 0) + e + ((e & 8) ? 8 : 0);
}

// ---------------------------------------------------------------------------
// Generic GEMM: Y[(n*M+m)<<lgV + v] = sum_k W[m*K+k] * X[((n*K+k)<<lgV) + v],
// j = (n,v), V = 1<<lgV (all voxel counts are powers of two -> shifts only).
// Workgroup tile 64(M) x 64(J); 8 waves; 2 WMMAs / k-step per wave.
// Staging: clamped addresses + float-mask multiply (straight-line, no exec).
// dynMode=1: per-z offsets for the per-(head,sample) dynamic conv.
// ---------------------------------------------------------------------------
__global__ __launch_bounds__(256) void k_gemm(
    const float* __restrict__ W, const float* __restrict__ X, float* __restrict__ Y,
    int M, int K, int Jtot, int lgV,
    long sA, long sB, long sC, int dynMode)
{
    __shared__ bf16_t As[64 * 32];   // [m][k]
    __shared__ bf16_t Bs[64 * 32];   // [j][k]

    const int tid  = threadIdx.x;
    const int lane = tid & 31;
    const int wave = tid >> 5;
    const int mw   = wave & 3;   // 4 m sub-tiles of 16
    const int jw   = wave >> 2;  // 2 j sub-tiles of 32 (two fragments each)
    const int Vm1  = (1 << lgV) - 1;

    const long z = blockIdx.z;
    const float* Wz; const float* Xz; float* Yz;
    if (dynMode) {
        int n = (int)z & 63, h = (int)z >> 6;            // z = h*64 + n
        Wz = W + ((long)(h * 64 + n)) * 512 * 256;       // w_dyn[h][n] : 512x256
        Xz = X + (long)n * 256 * 64;                     // h_act[n]   : 256x64
        Yz = Y + ((long)n * 3072 + (long)h * 512) * 64;  // lk[n][h*512+o][v]
    } else {
        Wz = W + z * sA; Xz = X + z * sB; Yz = Y + z * sC;
    }

    const int j0 = blockIdx.x * 64;
    const int m0 = blockIdx.y * 64;

    // staging: each thread handles 8 consecutive k of one row (branchless)
    const int sm = tid >> 2;          // 0..63  (A row / B column j)
    const int sk = (tid & 3) * 8;     // 0,8,16,24

    const int   gmA  = m0 + sm;
    const float mA   = (gmA < M) ? 1.0f : 0.0f;           // row mask as float
    const float* srcA = Wz + (long)(gmA < M ? gmA : 0) * K;

    const int   gjB  = j0 + sm;
    const float mB   = (gjB < Jtot) ? 1.0f : 0.0f;
    const int   gjc  = (gjB < Jtot) ? gjB : (Jtot - 1);
    const int   nB   = gjc >> lgV;
    const int   vvB  = gjc & Vm1;
    const float* srcB = Xz + (((long)nB * K) << lgV) + vvB;

    v8f acc0 = {}, acc1 = {};
    for (int kk = 0; kk < K; kk += 32) {
        // ---- stage A (64m x 32k) ----
        {
            if (kk + 64 <= K) __builtin_prefetch(srcA + kk + sk + 32, 0, 0);
            float f[8];
            #pragma unroll
            for (int q = 0; q < 8; ++q) {
                int gk  = kk + sk + q;
                int gkc = gk < K ? gk : (K - 1);
                float msk = (gk < K) ? mA : 0.0f;   // select on mask, not load
                f[q] = srcA[gkc] * msk;             // load always issued
            }
            uint4 pk;
            pk.x = pack_bf2(f[0], f[1]); pk.y = pack_bf2(f[2], f[3]);
            pk.z = pack_bf2(f[4], f[5]); pk.w = pack_bf2(f[6], f[7]);
            *(uint4*)(&As[sm * 32 + sk]) = pk;      // one ds_store_b128
        }
        // ---- stage B (64j x 32k, stored [j][k]) ----
        {
            float f[8];
            #pragma unroll
            for (int q = 0; q < 8; ++q) {
                int gk  = kk + sk + q;
                int gkc = gk < K ? gk : (K - 1);
                float msk = (gk < K) ? mB : 0.0f;
                f[q] = srcB[(long)gkc << lgV] * msk;
            }
            uint4 pk;
            pk.x = pack_bf2(f[0], f[1]); pk.y = pack_bf2(f[2], f[3]);
            pk.z = pack_bf2(f[4], f[5]); pk.w = pack_bf2(f[6], f[7]);
            *(uint4*)(&Bs[sm * 32 + sk]) = pk;
        }
        __syncthreads();

        v16bf a, b0, b1;
        const int mr = (mw << 4) + (lane & 15);
        const int jr = (jw << 5) + (lane & 15);
        #pragma unroll
        for (int e = 0; e < 16; ++e) {
            int kf = kfrag(lane, e);
            a[e]  = As[(mr << 5) + kf];
            b0[e] = Bs[(jr << 5) + kf];
            b1[e] = Bs[((jr + 16) << 5) + kf];
        }
        acc0 = __builtin_amdgcn_wmma_f32_16x16x32_bf16(
                   false, a, false, b0, (short)0, acc0, false, false);
        acc1 = __builtin_amdgcn_wmma_f32_16x16x32_bf16(
                   false, a, false, b1, (short)0, acc1, false, false);
        __syncthreads();
    }

    int gjA = j0 + (jw << 5) + (lane & 15);
    if (gjA < Jtot) {
        int n = gjA >> lgV, vv = gjA & Vm1;
        #pragma unroll
        for (int r = 0; r < 8; ++r) {
            int gm = m0 + (mw << 4) + r + ((lane >> 4) << 3);
            if (gm < M) Yz[((long)(n * M + gm) << lgV) + vv] = acc0[r];
        }
    }
    int gjD = gjA + 16;
    if (gjD < Jtot) {
        int n = gjD >> lgV, vv = gjD & Vm1;
        #pragma unroll
        for (int r = 0; r < 8; ++r) {
            int gm = m0 + (mw << 4) + r + ((lane >> 4) << 3);
            if (gm < M) Yz[((long)(n * M + gm) << lgV) + vv] = acc1[r];
        }
    }
}

// ---------------------------------------------------------------------------
// Implicit-GEMM 3x3x3 conv, pad=1, stride in {1,2}. K = Cin*27.
// Din = 1<<lgDin, Dout = 1<<lgDout (all spatial dims are powers of two).
// ---------------------------------------------------------------------------
__global__ __launch_bounds__(256) void k_conv3(
    const float* __restrict__ W, const float* __restrict__ X, float* __restrict__ Y,
    int Cout, int Cin, int N, int lgDin, int lgDout, int stride)
{
    __shared__ bf16_t As[64 * 32];
    __shared__ bf16_t Bs[64 * 32];

    const int tid  = threadIdx.x;
    const int lane = tid & 31;
    const int wave = tid >> 5;
    const int mw   = wave & 3;
    const int jw   = wave >> 2;

    const int K     = Cin * 27;
    const int Din   = 1 << lgDin;
    const int Dout  = 1 << lgDout;
    const int Vm1   = (1 << (3 * lgDout)) - 1;
    const int Jtot  = N << (3 * lgDout);
    const int j0 = blockIdx.x * 64;
    const int m0 = blockIdx.y * 64;

    const int sm = tid >> 2;
    const int sk = (tid & 3) * 8;

    const int   gmA = m0 + sm;
    const float mA  = (gmA < Cout) ? 1.0f : 0.0f;
    const float* srcA = W + (long)(gmA < Cout ? gmA : 0) * K;

    // hoist im2col j-decode (fixed per thread)
    const int   gjs = j0 + sm;
    const float mB  = (gjs < Jtot) ? 1.0f : 0.0f;
    const int   gjc = (gjs < Jtot) ? gjs : (Jtot - 1);
    const int   n_s = gjc >> (3 * lgDout);
    const int   r_s = gjc & Vm1;
    const int   od = r_s >> (2 * lgDout), oh = (r_s >> lgDout) & (Dout - 1),
                ow = r_s & (Dout - 1);
    const int   idb = od * stride - 1, ihb = oh * stride - 1, iwb = ow * stride - 1;
    const float* Xn = X + (((long)n_s * Cin) << (3 * lgDin));

    v8f acc0 = {}, acc1 = {};
    for (int kk = 0; kk < K; kk += 32) {
        // ---- stage A ----
        {
            if (kk + 64 <= K) __builtin_prefetch(srcA + kk + sk + 32, 0, 0);
            float f[8];
            #pragma unroll
            for (int q = 0; q < 8; ++q) {
                int gk  = kk + sk + q;
                int gkc = gk < K ? gk : (K - 1);
                float msk = (gk < K) ? mA : 0.0f;
                f[q] = srcA[gkc] * msk;
            }
            uint4 pk;
            pk.x = pack_bf2(f[0], f[1]); pk.y = pack_bf2(f[2], f[3]);
            pk.z = pack_bf2(f[4], f[5]); pk.w = pack_bf2(f[6], f[7]);
            *(uint4*)(&As[sm * 32 + sk]) = pk;
        }
        // ---- stage B (im2col gather, branchless) ----
        {
            float f[8];
            #pragma unroll
            for (int q = 0; q < 8; ++q) {
                int gk  = kk + sk + q;
                int gkc = gk < K ? gk : (K - 1);
                int ci = gkc / 27, t = gkc % 27;
                int id = idb + t / 9, ih = ihb + (t / 3) % 3, iw = iwb + t % 3;
                bool inb = (gk < K) &&
                           (unsigned)id < (unsigned)Din &&
                           (unsigned)ih < (unsigned)Din &&
                           (unsigned)iw < (unsigned)Din;
                float msk = inb ? mB : 0.0f;
                int idc = clampi(id, Din - 1), ihc = clampi(ih, Din - 1),
                    iwc = clampi(iw, Din - 1);
                long idx = ((((((long)ci << lgDin) + idc) << lgDin) + ihc) << lgDin) + iwc;
                f[q] = Xn[idx] * msk;
            }
            uint4 pk;
            pk.x = pack_bf2(f[0], f[1]); pk.y = pack_bf2(f[2], f[3]);
            pk.z = pack_bf2(f[4], f[5]); pk.w = pack_bf2(f[6], f[7]);
            *(uint4*)(&Bs[sm * 32 + sk]) = pk;
        }
        __syncthreads();

        v16bf a, b0, b1;
        const int mr = (mw << 4) + (lane & 15);
        const int jr = (jw << 5) + (lane & 15);
        #pragma unroll
        for (int e = 0; e < 16; ++e) {
            int kf = kfrag(lane, e);
            a[e]  = As[(mr << 5) + kf];
            b0[e] = Bs[(jr << 5) + kf];
            b1[e] = Bs[((jr + 16) << 5) + kf];
        }
        acc0 = __builtin_amdgcn_wmma_f32_16x16x32_bf16(
                   false, a, false, b0, (short)0, acc0, false, false);
        acc1 = __builtin_amdgcn_wmma_f32_16x16x32_bf16(
                   false, a, false, b1, (short)0, acc1, false, false);
        __syncthreads();
    }

    int gjA = j0 + (jw << 5) + (lane & 15);
    if (gjA < Jtot) {
        int n = gjA >> (3 * lgDout), r = gjA & Vm1;
        #pragma unroll
        for (int rr = 0; rr < 8; ++rr) {
            int gm = m0 + (mw << 4) + rr + ((lane >> 4) << 3);
            if (gm < Cout) Y[((long)(n * Cout + gm) << (3 * lgDout)) + r] = acc0[rr];
        }
    }
    int gjD = gjA + 16;
    if (gjD < Jtot) {
        int n = gjD >> (3 * lgDout), r = gjD & Vm1;
        #pragma unroll
        for (int rr = 0; rr < 8; ++rr) {
            int gm = m0 + (mw << 4) + rr + ((lane >> 4) << 3);
            if (gm < Cout) Y[((long)(n * Cout + gm) << (3 * lgDout)) + r] = acc1[rr];
        }
    }
}

// ---------------------------------------------------------------------------
// BatchNorm (train mode, batch stats, identity affine).  V = 1<<lgV.
// ---------------------------------------------------------------------------
__global__ __launch_bounds__(256) void k_bn_stats(
    const float* __restrict__ X, float* __restrict__ meanO, float* __restrict__ rstdO,
    int N, int C, int lgV)
{
    const int c = blockIdx.x;
    const int NV = N << lgV;
    const int Vm1 = (1 << lgV) - 1;
    float s = 0.f, sq = 0.f;
    for (int i = threadIdx.x; i < NV; i += 256) {
        int n = i >> lgV, v = i & Vm1;
        float x = X[((n * C + c) << lgV) + v];
        s += x; sq += x * x;
    }
    __shared__ float rs[256], rq[256];
    rs[threadIdx.x] = s; rq[threadIdx.x] = sq;
    __syncthreads();
    for (int o = 128; o > 0; o >>= 1) {
        if (threadIdx.x < o) { rs[threadIdx.x] += rs[threadIdx.x + o];
                               rq[threadIdx.x] += rq[threadIdx.x + o]; }
        __syncthreads();
    }
    if (threadIdx.x == 0) {
        float m = rs[0] / (float)NV;
        float var = rq[0] / (float)NV - m * m;
        meanO[c] = m;
        rstdO[c] = rsqrtf(var + 1e-5f);
    }
}

__global__ void k_bn_apply(float* __restrict__ X,
                           const float* __restrict__ mean, const float* __restrict__ rstd,
                           int C, int lgV, int total, int relu)
{
    int i = blockIdx.x * blockDim.x + threadIdx.x;
    if (i >= total) return;
    int c = (i >> lgV) % C;
    float x = (X[i] - mean[c]) * rstd[c];
    if (relu) x = fmaxf(x, 0.0f);
    X[i] = x;
}

__global__ void k_pool2(const float* __restrict__ X, float* __restrict__ Y,
                        int lgDin, int total)
{
    int i = blockIdx.x * blockDim.x + threadIdx.x;
    if (i >= total) return;
    const int lgDo = lgDin - 1;
    const int Dom1 = (1 << lgDo) - 1;
    int nc = i >> (3 * lgDo), r = i & ((1 << (3 * lgDo)) - 1);
    int od = r >> (2 * lgDo), oh = (r >> lgDo) & Dom1, ow = r & Dom1;
    const float* Xb = X + ((long)nc << (3 * lgDin));
    float s = 0.f;
    #pragma unroll
    for (int dz = 0; dz < 2; ++dz)
    #pragma unroll
    for (int dy = 0; dy < 2; ++dy)
    #pragma unroll
    for (int dx = 0; dx < 2; ++dx)
        s += Xb[((((od * 2 + dz) << lgDin) + (oh * 2 + dy)) << lgDin) + (ow * 2 + dx)];
    Y[i] = s * 0.125f;
}

__global__ void k_add2(const float* __restrict__ a, const float* __restrict__ b,
                       float* __restrict__ o, int total)
{
    int i = blockIdx.x * blockDim.x + threadIdx.x;
    if (i < total) o[i] = a[i] + b[i];
}

__global__ void k_segsum(const float* __restrict__ lk, const int* __restrict__ seg,
                         float* __restrict__ g, int CV, int Nn, int total)
{
    int i = blockIdx.x * blockDim.x + threadIdx.x;
    if (i >= total) return;
    int b = i / CV; int cv = i - b * CV;
    float s = 0.f;
    for (int n = 0; n < Nn; ++n)
        if (seg[n] == b) s += lk[n * CV + cv];
    g[i] = s;
}

// ---------------------------------------------------------------------------
extern "C" void kernel_launch(void* const* d_in, const int* in_sizes, int n_in,
                              void* d_out, int out_size, void* d_ws, size_t ws_size,
                              hipStream_t stream)
{
    (void)in_sizes; (void)n_in; (void)out_size; (void)ws_size;

    const float* x      = (const float*)d_in[0];
    const float* yv     = (const float*)d_in[1];
    const int*   seg    = (const int*)d_in[2];
    const float* tables = (const float*)d_in[15];

    struct RBp { const float *w1, *w2, *w3, *proj; int cin, bnc, cout; };
    RBp h1b[3] = {
        {(const float*)d_in[3],  (const float*)d_in[4],  (const float*)d_in[5],  (const float*)d_in[6],    1,   1,   64},
        {(const float*)d_in[7],  (const float*)d_in[8],  (const float*)d_in[9],  (const float*)d_in[10],  64,  16,  128},
        {(const float*)d_in[11], (const float*)d_in[12], (const float*)d_in[13], (const float*)d_in[14], 128,  32,  256},
    };
    RBp h2b[4] = {
        {(const float*)d_in[16], (const float*)d_in[17], (const float*)d_in[18], (const float*)d_in[19], 3072, 768, 1024},
        {(const float*)d_in[20], (const float*)d_in[21], (const float*)d_in[22], nullptr,                1024, 256, 1024},
        {(const float*)d_in[23], (const float*)d_in[24], (const float*)d_in[25], nullptr,                1024, 256, 1024},
        {(const float*)d_in[26], (const float*)d_in[27], (const float*)d_in[28], nullptr,                1024, 256, 1024},
    };

    float* p   = (float*)d_ws;
    float* stM = p;
    float* stR = p + 4096;
    const long CAP = 17000000;
    float* A  = p + 16384;
    float* Bb = A  + CAP;
    float* Cc = Bb + CAP;
    float* Dd = Cc + CAP;
    float* Ee = Dd + CAP;
    float* WD = A;    // 50.33M floats, reuses A..Cc span (dead then)
    float* LK = Dd;   // 12.58M floats

    auto gemm = [&](const float* W, const float* X, float* Y, int M, int K, int J, int lgV,
                    int Z, long sA, long sB, long sC, int dyn) {
        dim3 g((unsigned)((J + 63) / 64), (unsigned)((M + 63) / 64), (unsigned)Z);
        k_gemm<<<g, 256, 0, stream>>>(W, X, Y, M, K, J, lgV, sA, sB, sC, dyn);
    };
    auto conv3 = [&](const float* W, const float* X, float* Y,
                     int Cout, int Cin, int N, int lgDin, int stride) {
        int lgDout = (stride == 2) ? lgDin - 1 : lgDin;
        int J = N << (3 * lgDout);
        dim3 g((unsigned)((J + 63) / 64), (unsigned)((Cout + 63) / 64), 1);
        k_conv3<<<g, 256, 0, stream>>>(W, X, Y, Cout, Cin, N, lgDin, lgDout, stride);
    };
    auto bn = [&](float* X, int N, int C, int lgV, int relu) {
        k_bn_stats<<<C, 256, 0, stream>>>(X, stM, stR, N, C, lgV);
        int tot = (N * C) << lgV;
        k_bn_apply<<<(unsigned)((tot + 255) / 256), 256, 0, stream>>>(X, stM, stR, C, lgV, tot, relu);
    };
    auto pool = [&](const float* X, float* Y, int N, int C, int lgDin) {
        int tot = (N * C) << (3 * (lgDin - 1));
        k_pool2<<<(unsigned)((tot + 255) / 256), 256, 0, stream>>>(X, Y, lgDin, tot);
    };
    auto resblock = [&](const float* xin, int N, int lgDin, const RBp& rb,
                        int stride, int act_last,
                        float* out, float* t1, float* t2, float* t3) {
        int lgVin  = 3 * lgDin;
        int lgDout = (stride == 2) ? lgDin - 1 : lgDin;
        int lgVout = 3 * lgDout;
        gemm(rb.w1, xin, t1, rb.bnc, rb.cin, N << lgVin, lgVin, 1, 0, 0, 0, 0);
        bn(t1, N, rb.bnc, lgVin, 1);
        conv3(rb.w2, t1, t2, rb.bnc, rb.bnc, N, lgDin, stride);
        bn(t2, N, rb.bnc, lgVout, 1);
        gemm(rb.w3, t2, t3, rb.cout, rb.bnc, N << lgVout, lgVout, 1, 0, 0, 0, 0);
        bn(t3, N, rb.cout, lgVout, act_last);
        const float* res = xin;
        if (stride == 2) { pool(xin, t1, N, rb.cin, lgDin); res = t1; }
        if (rb.proj) {
            gemm(rb.proj, res, t2, rb.cout, rb.cin, N << lgVout, lgVout, 1, 0, 0, 0, 0);
            res = t2;
        }
        int tot = (N * rb.cout) << lgVout;
        k_add2<<<(unsigned)((tot + 255) / 256), 256, 0, stream>>>(t3, res, out, tot);
    };

    // ---- head_1 : [64,1,32^3] -> [64,256,4^3] ----
    resblock(x,  64, 5, h1b[0], 2, 1, A,  Bb, Cc, Dd);   // -> A  [64,64,16^3]
    resblock(A,  64, 4, h1b[1], 2, 1, Bb, Cc, Dd, Ee);   // -> Bb [64,128,8^3]
    resblock(Bb, 64, 3, h1b[2], 2, 1, Ee, A,  Cc, Dd);   // -> Ee [64,256,4^3]

    // ---- dynamic per-sample conv ----
    // w_dyn[h][n][f] = sum_q y[n][q] * tables[h][q][f]   (WMMA GEMM, Z = heads)
    gemm(yv, tables, WD, 64, 24, 131072, 17, 6,
         0, 24L * 131072, 64L * 131072, 0);
    // lk[n][h*512+o][v] = sum_c h[n][c][v] * w_dyn[h][n][o][c]  (384 WMMA GEMMs)
    gemm(WD, Ee, LK, 512, 256, 64, 6, 384, 0, 0, 0, /*dynMode=*/1);

    // ---- ragged segment sum -> g [8,3072,4^3], then BN+ReLU ----
    {
        int CV = 3072 * 64, tot = 8 * CV;
        k_segsum<<<(unsigned)((tot + 255) / 256), 256, 0, stream>>>(LK, seg, A, CV, 64, tot);
        bn(A, 8, 3072, 6, 1);
    }

    // ---- head_2 : [8,3072,4^3] -> [8,1024,1] ----
    resblock(A,  8, 2, h2b[0], 2, 1, Bb, Cc, Dd, Ee);
    resblock(Bb, 8, 1, h2b[1], 2, 1, A,  Cc, Dd, Ee);
    resblock(A,  8, 0, h2b[2], 1, 1, Bb, Cc, Dd, Ee);
    resblock(Bb, 8, 0, h2b[3], 1, 0, (float*)d_out, Cc, Dd, Ee);
}
